// MSGSetConv_48163763257798
// MI455X (gfx1250) — matrix operand order repaired and checked by reading, not compiled, not run on hardware
//
#include <hip/hip_runtime.h>
#include <math.h>

typedef __bf16 bf16_t;
typedef __bf16 v16bf __attribute__((ext_vector_type(16)));
typedef __bf16 v8bf  __attribute__((ext_vector_type(8)));
typedef float  v8f   __attribute__((ext_vector_type(8)));

constexpr int B_ = 16, N_ = 4096, C_ = 64, S_ = 1024;

// ---------------------------------------------------------------------------
// Furthest point sampling: one block per batch, 1024 threads (32 waves).
// Each thread owns 4 points; per-round wave shfl-xor argmax + LDS cross-wave.
// Writes new_xyz (B,S,3) directly into d_out.
// ---------------------------------------------------------------------------
__global__ __launch_bounds__(1024) void fps_kernel(const float* __restrict__ xyz,
                                                   float* __restrict__ nxyz)
{
  const int b = blockIdx.x;
  const int tid = threadIdx.x;
  __shared__ float bx, by, bz;
  __shared__ float wval[32];
  __shared__ int   widx[32];

  float px[4], py[4], pz[4], dmin[4];
#pragma unroll
  for (int j = 0; j < 4; ++j) {
    int p = j * 1024 + tid;
    const float* q = xyz + ((size_t)b * N_ + p) * 3;
    px[j] = q[0]; py[j] = q[1]; pz[j] = q[2];
    dmin[j] = 1e10f;
  }
  if (tid == 0) {
    const float* q = xyz + (size_t)b * N_ * 3;
    bx = q[0]; by = q[1]; bz = q[2];
    float* o = nxyz + (size_t)b * S_ * 3;
    o[0] = q[0]; o[1] = q[1]; o[2] = q[2];
  }
  __syncthreads();

  for (int i = 1; i < S_; ++i) {
    float lx = bx, ly = by, lz = bz;
    float best = -1.f; int bidx = 0;
#pragma unroll
    for (int j = 0; j < 4; ++j) {
      float dx = px[j] - lx, dy = py[j] - ly, dz = pz[j] - lz;
      float d = dx * dx + dy * dy + dz * dz;
      dmin[j] = fminf(dmin[j], d);
      if (dmin[j] > best) { best = dmin[j]; bidx = j * 1024 + tid; }
    }
#pragma unroll
    for (int o = 1; o < 32; o <<= 1) {
      float ov = __shfl_xor(best, o, 32);
      int   oi = __shfl_xor(bidx, o, 32);
      if (ov > best || (ov == best && oi < bidx)) { best = ov; bidx = oi; }
    }
    if ((tid & 31) == 0) { wval[tid >> 5] = best; widx[tid >> 5] = bidx; }
    __syncthreads();
    if (tid < 32) {
      best = wval[tid]; bidx = widx[tid];
#pragma unroll
      for (int o = 1; o < 32; o <<= 1) {
        float ov = __shfl_xor(best, o, 32);
        int   oi = __shfl_xor(bidx, o, 32);
        if (ov > best || (ov == best && oi < bidx)) { best = ov; bidx = oi; }
      }
      if (tid == 0) {
        const float* q = xyz + ((size_t)b * N_ + bidx) * 3;
        float sx = q[0], sy = q[1], sz = q[2];
        bx = sx; by = sy; bz = sz;
        float* o2 = nxyz + ((size_t)b * S_ + i) * 3;
        o2[0] = sx; o2[1] = sy; o2[2] = sz;
      }
    }
    __syncthreads();
  }
}

// ---------------------------------------------------------------------------
// features (B,C,N) f32 -> feat_t (B,N,C) bf16 (128B per point, L2-resident)
// ---------------------------------------------------------------------------
__global__ __launch_bounds__(256) void feat_conv_kernel(const float* __restrict__ f,
                                                        bf16_t* __restrict__ ft)
{
  size_t i = (size_t)blockIdx.x * 256 + threadIdx.x;   // over B*C*N
  int n = (int)(i & (N_ - 1));
  int c = (int)((i >> 12) & (C_ - 1));
  int b = (int)(i >> 18);
  ft[((size_t)b * N_ + n) * C_ + c] = (bf16_t)f[i];
}

// ---------------------------------------------------------------------------
// Weight prep: permute layer-1 channels to [feat(64), xyz(3)], pad K to 96,
// convert to bf16 row-major; fold BN into per-channel scale/bias.
// ---------------------------------------------------------------------------
__global__ __launch_bounds__(256) void wprep_kernel(
    const float* __restrict__ w, const float* __restrict__ g,
    const float* __restrict__ be, const float* __restrict__ mn,
    const float* __restrict__ vr,
    bf16_t* __restrict__ wp, float* __restrict__ sb,
    int cout, int cin, int xyzfirst)
{
  int t = blockIdx.x * 256 + threadIdx.x;
  if (t >= cout * 96) return;
  int o = t / 96, k = t - o * 96;
  float val;
  if (xyzfirst) val = (k < 64) ? w[o * cin + 3 + k] : ((k < 67) ? w[o * cin + (k - 64)] : 0.f);
  else          val = (k < cin) ? w[o * cin + k] : 0.f;
  wp[o * 96 + k] = (bf16_t)val;
  if (k == 0) {
    float s = g[o] * rsqrtf(vr[o] + 1e-5f);
    sb[o] = s;
    sb[cout + o] = be[o] - mn[o] * s;
  }
}

// ---------------------------------------------------------------------------
// Ball query: one wave per center; wave32 ballot ordered compaction keeps
// first-in-index-order hits; pads with first hit (matches reference).
// ---------------------------------------------------------------------------
__global__ __launch_bounds__(256) void bq_kernel(const float* __restrict__ xyz,
                                                 const float* __restrict__ nxyz,
                                                 int* __restrict__ idxo,
                                                 float r2, int ns)
{
  const int lane = threadIdx.x & 31;
  const int cid  = blockIdx.x * 8 + (threadIdx.x >> 5);
  const int b = cid >> 10;
  const float* c = nxyz + (size_t)cid * 3;
  const float cx = c[0], cy = c[1], cz = c[2];
  const float* base = xyz + (size_t)b * N_ * 3;
  int* ow = idxo + (size_t)cid * ns;
  int cnt = 0, first = 0;
  for (int p0 = 0; p0 < N_; p0 += 32) {
    int p = p0 + lane;
    const float* q = base + (size_t)p * 3;
    float dx = q[0] - cx, dy = q[1] - cy, dz = q[2] - cz;
    bool hit = (dx * dx + dy * dy + dz * dz) < r2;
    unsigned m = (unsigned)__ballot(hit);
    if (cnt == 0 && m) first = p0 + __builtin_ctz(m);
    int pre = __popc(m & ((1u << lane) - 1u));
    if (hit && (cnt + pre) < ns) ow[cnt + pre] = p;
    cnt += __popc(m);
    if (cnt >= ns) break;
  }
  for (int i = cnt + lane; i < ns; i += 32) ow[i] = first;
}

// ---------------------------------------------------------------------------
// One MLP layer via v_wmma_f32_16x16x32_bf16, fused BN+ReLU.
// K-outer loop: B (activation) fragment loaded ONCE per K-step from LDS and
// reused across all M-tiles (accumulators held live per M-tile).
// ---------------------------------------------------------------------------
template<int COUT, int KD, bool LAST>
__device__ __forceinline__ void mlp_layer(
    const bf16_t* __restrict__ wp, const float* __restrict__ sb,
    const bf16_t* src, bf16_t* dst, float* vmax, int col, int half)
{
  constexpr int MT = COUT / 16;
  v8f acc[MT];
#pragma unroll
  for (int m = 0; m < MT; ++m)
#pragma unroll
    for (int r = 0; r < 8; ++r) acc[m][r] = 0.f;

#pragma unroll
  for (int k = 0; k < KD / 32; ++k) {
    // B fragment: lane holds column `col`, 16 consecutive K at half*16.
    const bf16_t* xr = src + col * 96 + k * 32 + half * 16;
    v8bf b0 = *(const v8bf*)(xr);
    v8bf b1 = *(const v8bf*)(xr + 8);
    v16bf Bv = __builtin_shufflevector(b0, b1, 0,1,2,3,4,5,6,7,8,9,10,11,12,13,14,15);
#pragma unroll
    for (int m = 0; m < MT; ++m) {
      // A fragment: lane holds row M=col of this tile; K split by lane half.
      const bf16_t* wr = wp + (size_t)(m * 16 + col) * 96 + k * 32 + half * 16;
      v8bf a0 = *(const v8bf*)(wr);
      v8bf a1 = *(const v8bf*)(wr + 8);
      v16bf A = __builtin_shufflevector(a0, a1, 0,1,2,3,4,5,6,7,8,9,10,11,12,13,14,15);
      acc[m] = __builtin_amdgcn_wmma_f32_16x16x32_bf16(false, A, false, Bv,
                                                       (short)0, acc[m], false, false);
    }
  }

#pragma unroll
  for (int m = 0; m < MT; ++m) {
    const float* sc = sb + m * 16 + half * 8;
    const float* bi = sb + COUT + m * 16 + half * 8;
    if constexpr (LAST) {
#pragma unroll
      for (int r = 0; r < 8; ++r) {
        float v = fmaxf(acc[m][r] * sc[r] + bi[r], 0.f);
        vmax[m * 8 + r] = fmaxf(vmax[m * 8 + r], v);
      }
    } else {
      v8bf p;
#pragma unroll
      for (int r = 0; r < 8; ++r) p[r] = (bf16_t)fmaxf(acc[m][r] * sc[r] + bi[r], 0.f);
      *(v8bf*)(dst + col * 96 + m * 16 + half * 8) = p;
    }
  }
}

// ---------------------------------------------------------------------------
// Fused gather + 3-layer MLP + neighbor max-pool. One wave per center;
// 16-neighbor column tiles stream through two ping-pong LDS buffers.
// ---------------------------------------------------------------------------
template<int C1, int C2, int C3, int NS, int COFF>
__global__ __launch_bounds__(256) void msg_kernel(
    const bf16_t* __restrict__ ft, const float* __restrict__ xyz,
    const float* __restrict__ nxyz, const int* __restrict__ idxp,
    const bf16_t* __restrict__ wp1, const bf16_t* __restrict__ wp2,
    const bf16_t* __restrict__ wp3,
    const float* __restrict__ sb1, const float* __restrict__ sb2,
    const float* __restrict__ sb3,
    float* __restrict__ outp)
{
  __shared__ bf16_t smem[8 * 2 * 1536];      // 48 KB: 8 waves x 2 x (96x16)
  const int w = threadIdx.x >> 5, lane = threadIdx.x & 31;
  const int col = lane & 15, half = lane >> 4;
  const int cid = blockIdx.x * 8 + w;
  const int b = cid >> 10, s = cid & (S_ - 1);
  bf16_t* bufX = smem + w * 3072;
  bf16_t* bufH = bufX + 1536;

  const float* cp = nxyz + (size_t)cid * 3;
  const float cx = cp[0], cy = cp[1], cz = cp[2];

  float vmax[C3 / 2];
#pragma unroll
  for (int i = 0; i < C3 / 2; ++i) vmax[i] = 0.f;

  v8bf zero8;
#pragma unroll
  for (int i = 0; i < 8; ++i) zero8[i] = (bf16_t)0.f;

  const int* ip = idxp + (size_t)cid * NS;

  for (int t = 0; t < NS / 16; ++t) {
    // ---- gather tile: rows [0,64)=features, [64,67)=centered xyz, rest 0 ---
    int n = ip[t * 16 + col];
    const bf16_t* fp = ft + ((size_t)b * N_ + n) * 64 + half * 32;
    v8bf f0 = *(const v8bf*)(fp + 0);
    v8bf f1 = *(const v8bf*)(fp + 8);
    v8bf f2 = *(const v8bf*)(fp + 16);
    v8bf f3 = *(const v8bf*)(fp + 24);
    bf16_t* xc = bufX + col * 96;
    *(v8bf*)(xc + half * 32 + 0)  = f0;
    *(v8bf*)(xc + half * 32 + 8)  = f1;
    *(v8bf*)(xc + half * 32 + 16) = f2;
    *(v8bf*)(xc + half * 32 + 24) = f3;
    if (half == 0) {
      const float* q = xyz + ((size_t)b * N_ + n) * 3;
      v8bf t0 = zero8;
      t0[0] = (bf16_t)(q[0] - cx);
      t0[1] = (bf16_t)(q[1] - cy);
      t0[2] = (bf16_t)(q[2] - cz);
      *(v8bf*)(xc + 64) = t0;
      *(v8bf*)(xc + 72) = zero8;
    } else {
      *(v8bf*)(xc + 80) = zero8;
      *(v8bf*)(xc + 88) = zero8;
    }
    // ---- 3-layer MLP, LDS ping-pong: X->H1->X(=H2)->max -------------------
    mlp_layer<C1, 96, false>(wp1, sb1, bufX, bufH, nullptr, col, half);
    mlp_layer<C2, C1, false>(wp2, sb2, bufH, bufX, nullptr, col, half);
    mlp_layer<C3, C2, true >(wp3, sb3, bufX, nullptr, vmax, col, half);
  }

  // ---- max over the 16 columns each half holds, then write (B,320,S) ------
#pragma unroll
  for (int m = 0; m < C3 / 16; ++m) {
#pragma unroll
    for (int r = 0; r < 8; ++r) {
      float v = vmax[m * 8 + r];
#pragma unroll
      for (int o = 1; o < 16; o <<= 1) v = fmaxf(v, __shfl_xor(v, o, 32));
      if (col == 0)
        outp[((size_t)b * 320 + COFF + m * 16 + half * 8 + r) * S_ + s] = v;
    }
  }
}

// ---------------------------------------------------------------------------
extern "C" void kernel_launch(void* const* d_in, const int* in_sizes, int n_in,
                              void* d_out, int out_size, void* d_ws, size_t ws_size,
                              hipStream_t stream)
{
  (void)in_sizes; (void)n_in; (void)out_size; (void)ws_size;
  const float* xyz  = (const float*)d_in[0];
  const float* feat = (const float*)d_in[1];

  float* out  = (float*)d_out;
  float* nxyz = out;                              // (B,S,3)
  float* fout = out + (size_t)B_ * S_ * 3;        // (B,320,S)

  char* ws = (char*)d_ws;
  size_t off = 0;
  bf16_t* ft = (bf16_t*)(ws + off); off += (size_t)B_ * N_ * C_ * 2;   // 8 MB
  int* idx0 = (int*)(ws + off); off += (size_t)B_ * S_ * 16 * 4;
  int* idx1 = (int*)(ws + off); off += (size_t)B_ * S_ * 32 * 4;
  int* idx2 = (int*)(ws + off); off += (size_t)B_ * S_ * 64 * 4;

  static const int COUTS[9] = {32, 32, 64, 64, 64, 128, 64, 96, 128};
  static const int CINS[9]  = {67, 32, 32, 67, 64, 64, 67, 64, 96};
  bf16_t* wp[9]; float* sb[9];
  for (int l = 0; l < 9; ++l) {
    wp[l] = (bf16_t*)(ws + off); off += (size_t)COUTS[l] * 96 * 2;
    sb[l] = (float*)(ws + off);  off += (size_t)COUTS[l] * 2 * 4;
  }

  fps_kernel<<<B_, 1024, 0, stream>>>(xyz, nxyz);
  feat_conv_kernel<<<(B_ * C_ * N_) / 256, 256, 0, stream>>>(feat, ft);
  for (int l = 0; l < 9; ++l) {
    const float* w  = (const float*)d_in[2 + l * 5 + 0];
    const float* g  = (const float*)d_in[2 + l * 5 + 1];
    const float* be = (const float*)d_in[2 + l * 5 + 2];
    const float* mn = (const float*)d_in[2 + l * 5 + 3];
    const float* vr = (const float*)d_in[2 + l * 5 + 4];
    wprep_kernel<<<(COUTS[l] * 96 + 255) / 256, 256, 0, stream>>>(
        w, g, be, mn, vr, wp[l], sb[l], COUTS[l], CINS[l], (l % 3 == 0) ? 1 : 0);
  }
  bq_kernel<<<(B_ * S_) / 8, 256, 0, stream>>>(xyz, nxyz, idx0, 0.01f, 16);
  bq_kernel<<<(B_ * S_) / 8, 256, 0, stream>>>(xyz, nxyz, idx1, 0.04f, 32);
  bq_kernel<<<(B_ * S_) / 8, 256, 0, stream>>>(xyz, nxyz, idx2, 0.16f, 64);

  msg_kernel<32, 32, 64, 16, 0><<<(B_ * S_) / 8, 256, 0, stream>>>(
      ft, xyz, nxyz, idx0, wp[0], wp[1], wp[2], sb[0], sb[1], sb[2], fout);
  msg_kernel<64, 64, 128, 32, 64><<<(B_ * S_) / 8, 256, 0, stream>>>(
      ft, xyz, nxyz, idx1, wp[3], wp[4], wp[5], sb[3], sb[4], sb[5], fout);
  msg_kernel<64, 96, 128, 64, 192><<<(B_ * S_) / 8, 256, 0, stream>>>(
      ft, xyz, nxyz, idx2, wp[6], wp[7], wp[8], sb[6], sb[7], sb[8], fout);
}
